// RNNSeq2Seq_87780541596097
// MI455X (gfx1250) — matrix operand order repaired and checked by reading, not compiled, not run on hardware
//
#include <hip/hip_runtime.h>
#include <hip/hip_bf16.h>

// ---------------------------------------------------------------------------
// Types for CDNA5 WMMA
// ---------------------------------------------------------------------------
typedef __attribute__((ext_vector_type(16))) __bf16 v16bf;
typedef __attribute__((ext_vector_type(8)))  float  v8f;
typedef __attribute__((ext_vector_type(4)))  float  f32x4;

static constexpr int Bsz  = 64;
static constexpr int Tlen = 512;
static constexpr int HIDc = 256;
static constexpr int H2c  = 512;
static constexpr int VOCc = 32000;

// ---------------------------------------------------------------------------
// Weight pre-pass: convert f32 row-major W[K,N] -> bf16 column-major Wt[N,K].
// Done once per launch; weights are then streamed from L2 as contiguous
// 32-byte-per-lane fragments in the GEMM hot loop (no cvt, no strided b32s).
// ---------------------------------------------------------------------------
__global__ __launch_bounds__(256) void convert_wt_kernel(
    const float* __restrict__ W, __bf16* __restrict__ Wt, int K, int N)
{
  int i = (int)(blockIdx.x * blockDim.x + threadIdx.x);   // grid == K*N
  int n = i / K;
  int k = i - n * K;
  Wt[i] = (__bf16)W[(size_t)k * N + n];
}

// ---------------------------------------------------------------------------
// bf16-input, f32-accumulate WMMA GEMM:
//   Cout[M,N] = A[M,K] @ B[K,N]  (+ Cin[M,N])  (+ bias[N])
// B is pre-swizzled bf16 column-major (Bt[n*K + k]).
// Each wave computes a 16x64 output strip: one shared A fragment feeds
// 4 v_wmma_f32_16x16x32_bf16 per K-step. Software-pipelined (double-buffered)
// so the next K-step's loads overlap the current step's WMMAs.
// A/B lanes hold contiguous k (k = 16*half + s, not the ISA table order); C/D per CDNA5 ISA 7.12.2.
// ---------------------------------------------------------------------------
__global__ __launch_bounds__(256) void wmma_gemm_bf16(
    const float* __restrict__ A, int lda,
    const __bf16* __restrict__ Bt,
    const float* __restrict__ Cin, int ldcin,
    float* __restrict__ Cout, int ldc,
    const float* __restrict__ bias,
    int M, int N, int K)
{
  int gwave = (int)((blockIdx.x * blockDim.x + threadIdx.x) >> 5);
  int lane  = (int)(threadIdx.x & 31);
  int tnCnt = N >> 6;                 // 64-wide N strips
  int tm = gwave / tnCnt;
  int tn = gwave - tm * tnCnt;
  if (tm * 16 >= M) return;           // wave-uniform: EXEC stays all-ones

  int row0 = tm << 4;
  int col0 = tn << 6;
  int hs   = lane >> 4;               // half-select (0: lanes 0-15, 1: 16-31)
  int lr   = lane & 15;

  v8f acc[4];
#pragma unroll
  for (int j = 0; j < 4; ++j) {
    int col = col0 + j * 16 + lr;
    if (Cin) {
#pragma unroll
      for (int r = 0; r < 8; ++r)
        acc[j][r] = Cin[(size_t)(row0 + hs * 8 + r) * (size_t)ldcin + col];
    } else {
#pragma unroll
      for (int r = 0; r < 8; ++r) acc[j][r] = 0.0f;
    }
  }

  const float* arowBase = A + (size_t)(row0 + lr) * (size_t)lda + hs * 16;
  const __bf16* bBase[4];
#pragma unroll
  for (int j = 0; j < 4; ++j)
    bBase[j] = Bt + (size_t)(col0 + j * 16 + lr) * (size_t)K + hs * 16;

  // A tile 16x32: lane<16 -> row=lr, K = {k0..k0+15}
  //               lane>=16 -> row=lr, K = {k0+16..k0+31}
  auto loadA = [&](int k0) {
    const float* arow = arowBase + k0;
    f32x4 a0 = *(const f32x4*)(arow);
    f32x4 a1 = *(const f32x4*)(arow + 4);
    f32x4 a2 = *(const f32x4*)(arow + 8);
    f32x4 a3 = *(const f32x4*)(arow + 12);
    v16bf av;
#pragma unroll
    for (int e = 0; e < 4; ++e) {
      av[e]      = (__bf16)a0[e];
      av[e + 4]  = (__bf16)a1[e];
      av[e + 8]  = (__bf16)a2[e];
      av[e + 12] = (__bf16)a3[e];
    }
    return av;
  };
  // B tile 32x16: lanes 0-15 hold K=k0..k0+15 of its column, lanes 16-31
  // hold K=k0+16..k0+31; contiguous 32 bytes per lane (2x b128).
  auto loadB = [&](int j, int k0) { return *(const v16bf*)(bBase[j] + k0); };

  // ---- software-pipelined K loop (K >= 64, multiple of 32) ----
  v16bf av = loadA(0);
  v16bf bv0 = loadB(0, 0), bv1 = loadB(1, 0), bv2 = loadB(2, 0),
        bv3 = loadB(3, 0);
  int k0 = 0;
  for (; k0 + 32 < K; k0 += 32) {
    int kn = k0 + 32;
    __builtin_prefetch(arowBase + kn + 32, 0, 1);   // global_prefetch_b8
    v16bf avN  = loadA(kn);
    v16bf bv0N = loadB(0, kn);
    v16bf bv1N = loadB(1, kn);
    v16bf bv2N = loadB(2, kn);
    v16bf bv3N = loadB(3, kn);
    acc[0] = __builtin_amdgcn_wmma_f32_16x16x32_bf16(false, av, false, bv0,
                                                     (short)0, acc[0], false, false);
    acc[1] = __builtin_amdgcn_wmma_f32_16x16x32_bf16(false, av, false, bv1,
                                                     (short)0, acc[1], false, false);
    acc[2] = __builtin_amdgcn_wmma_f32_16x16x32_bf16(false, av, false, bv2,
                                                     (short)0, acc[2], false, false);
    acc[3] = __builtin_amdgcn_wmma_f32_16x16x32_bf16(false, av, false, bv3,
                                                     (short)0, acc[3], false, false);
    av = avN; bv0 = bv0N; bv1 = bv1N; bv2 = bv2N; bv3 = bv3N;
  }
  // epilogue: last K-step
  acc[0] = __builtin_amdgcn_wmma_f32_16x16x32_bf16(false, av, false, bv0,
                                                   (short)0, acc[0], false, false);
  acc[1] = __builtin_amdgcn_wmma_f32_16x16x32_bf16(false, av, false, bv1,
                                                   (short)0, acc[1], false, false);
  acc[2] = __builtin_amdgcn_wmma_f32_16x16x32_bf16(false, av, false, bv2,
                                                   (short)0, acc[2], false, false);
  acc[3] = __builtin_amdgcn_wmma_f32_16x16x32_bf16(false, av, false, bv3,
                                                   (short)0, acc[3], false, false);

#pragma unroll
  for (int j = 0; j < 4; ++j) {
    int col = col0 + j * 16 + lr;
    float bb = bias ? bias[col] : 0.0f;
#pragma unroll
    for (int r = 0; r < 8; ++r)
      Cout[(size_t)(row0 + hs * 8 + r) * (size_t)ldc + col] = acc[j][r] + bb;
  }
}

// ---------------------------------------------------------------------------
// Embedding gather + Keras Masking: one wave per (b,t) row of 256 floats.
// mask = any(emb[token] != 0); x = emb[token] * mask.
// ---------------------------------------------------------------------------
__global__ __launch_bounds__(256) void embed_mask_kernel(
    const int* __restrict__ tokens, const float* __restrict__ emb,
    float* __restrict__ x, float* __restrict__ mask, int nrows)
{
  int wid  = (int)((blockIdx.x * blockDim.x + threadIdx.x) >> 5);
  int lane = (int)(threadIdx.x & 31);
  if (wid >= nrows) return;
  int tok = tokens[wid];
  const float* e = emb + (size_t)tok * HIDc;
  float v[8];
  int nz = 0;
#pragma unroll
  for (int j = 0; j < 8; ++j) {
    v[j] = e[lane * 8 + j];
    nz |= (v[j] != 0.0f);
  }
  unsigned long long bal = __ballot(nz);
  float m = (bal != 0ull) ? 1.0f : 0.0f;
#pragma unroll
  for (int j = 0; j < 8; ++j)
    x[(size_t)wid * HIDc + lane * 8 + j] = v[j] * m;
  if (lane == 0) mask[wid] = m;
}

// ---------------------------------------------------------------------------
// LSTM gate pointwise + masked state update. Z is [B, 4H] (i,f,g,o order).
// y[(b*T + t)*ldy + ycol0 + j] = h2   (Keras: masked steps carry state).
// ---------------------------------------------------------------------------
__device__ __forceinline__ float sigmoidf(float x) {
  return 1.0f / (1.0f + __expf(-x));
}

__global__ __launch_bounds__(256) void lstm_pointwise_kernel(
    const float* __restrict__ Z, float* __restrict__ h, float* __restrict__ c,
    const float* __restrict__ mask, int t, int T,
    float* __restrict__ y, int ldy, int ycol0, int H)
{
  int idx = (int)(blockIdx.x * blockDim.x + threadIdx.x);   // grid == B*H
  int b = idx / H;
  int j = idx - b * H;
  const float* z = Z + (size_t)b * 4 * H;
  float zi = z[j], zf = z[H + j], zg = z[2 * H + j], zo = z[3 * H + j];
  float ig = sigmoidf(zi);
  float fg = sigmoidf(zf);
  float og = sigmoidf(zo);
  float gg = tanhf(zg);
  float cold = c[idx], hold = h[idx];
  float cn = fg * cold + ig * gg;
  float hn = og * tanhf(cn);
  float m = mask[(size_t)b * T + t];
  float h2 = (m > 0.5f) ? hn : hold;
  float c2 = (m > 0.5f) ? cn : cold;
  h[idx] = h2;
  c[idx] = c2;
  y[(size_t)(b * T + t) * (size_t)ldy + ycol0 + j] = h2;
}

// ---------------------------------------------------------------------------
// Small utility kernels
// ---------------------------------------------------------------------------
__global__ __launch_bounds__(256) void zero_kernel(float* __restrict__ p, int n)
{
  int i = (int)(blockIdx.x * blockDim.x + threadIdx.x);
  if (i < n) p[i] = 0.0f;
}

// dst[r, col0 + c] = src[r, c]   (src compact rows x cols, dst has ldd)
__global__ __launch_bounds__(256) void copy_cols_kernel(
    const float* __restrict__ src, float* __restrict__ dst,
    int rows, int cols, int ldd, int col0)
{
  int i = (int)(blockIdx.x * blockDim.x + threadIdx.x);
  if (i >= rows * cols) return;
  int r = i / cols;
  int cc = i - r * cols;
  dst[(size_t)r * ldd + col0 + cc] = src[i];
}

// ---------------------------------------------------------------------------
// Host orchestration
// ---------------------------------------------------------------------------
extern "C" void kernel_launch(void* const* d_in, const int* in_sizes, int n_in,
                              void* d_out, int out_size, void* d_ws, size_t ws_size,
                              hipStream_t stream) {
  (void)in_sizes; (void)n_in; (void)out_size; (void)ws_size;

  const int* enc_tokens = (const int*)d_in[0];
  const int* dec_tokens = (const int*)d_in[1];
  const float* emb      = (const float*)d_in[2];

  // enc_params: 2 layers x 2 dirs x (Wk, Wr, b)
  const float* encWk[2][2]; const float* encWr[2][2]; const float* encBb[2][2];
  int p = 3;
  for (int l = 0; l < 2; ++l)
    for (int d = 0; d < 2; ++d) {
      encWk[l][d] = (const float*)d_in[p++];
      encWr[l][d] = (const float*)d_in[p++];
      encBb[l][d] = (const float*)d_in[p++];
    }
  // dec_params: 2 layers x (Wk, Wr, b)
  const float* decWk[2]; const float* decWr[2]; const float* decBb[2];
  for (int l = 0; l < 2; ++l) {
    decWk[l] = (const float*)d_in[p++];
    decWr[l] = (const float*)d_in[p++];
    decBb[l] = (const float*)d_in[p++];
  }
  const float* dense_W = (const float*)d_in[p++];
  const float* dense_b = (const float*)d_in[p++];
  float* out = (float*)d_out;

  // ---- workspace layout ----
  float* ws = (float*)d_ws;
  const size_t ROWS = (size_t)Bsz * Tlen;            // 32768
  float* buf0  = ws;                ws += ROWS * 512;            // activations ping
  float* buf1  = ws;                ws += ROWS * 512;            // activations pong
  float* zxBuf = ws;                ws += ROWS * 2048;           // precomputed x@Wk + b
  float* zStep = ws;                ws += (size_t)Bsz * 2048;    // per-step gate pre-acts
  float* fh = ws;                   ws += (size_t)Bsz * HIDc;
  float* fc = ws;                   ws += (size_t)Bsz * HIDc;
  float* bh = ws;                   ws += (size_t)Bsz * HIDc;
  float* bc = ws;                   ws += (size_t)Bsz * HIDc;
  float* hdec = ws;                 ws += (size_t)Bsz * H2c;
  float* cdec = ws;                 ws += (size_t)Bsz * H2c;
  float* maskE = ws;                ws += ROWS;
  float* maskD = ws;                ws += ROWS;

  // bf16 column-major weight shadows (aligned: all prior counts are x16 floats)
  __bf16* wtb = (__bf16*)ws;
  size_t wo = 0;
  auto allocWt = [&](size_t elems) { __bf16* r = wtb + wo; wo += elems; return r; };

  __bf16* encWkT[2][2]; __bf16* encWrT[2][2];
  for (int l = 0; l < 2; ++l)
    for (int d = 0; d < 2; ++d) {
      int din = (l == 0) ? HIDc : 2 * HIDc;
      encWkT[l][d] = allocWt((size_t)din * 4 * HIDc);
      encWrT[l][d] = allocWt((size_t)HIDc * 4 * HIDc);
    }
  __bf16* decWkT[2]; __bf16* decWrT[2];
  for (int l = 0; l < 2; ++l) {
    int din = (l == 0) ? HIDc : H2c;
    decWkT[l] = allocWt((size_t)din * 4 * H2c);
    decWrT[l] = allocWt((size_t)H2c * 4 * H2c);
  }
  __bf16* denseWT = allocWt((size_t)H2c * VOCc);

  auto convert = [&](const float* W, __bf16* Wt, int K, int N) {
    convert_wt_kernel<<<(K * N) / 256, 256, 0, stream>>>(W, Wt, K, N);
  };
  auto gemm = [&](const float* A, int lda, const __bf16* Bt,
                  const float* Cin, int ldcin, float* C, int ldc,
                  const float* bias, int M, int N, int K) {
    int tiles = (M / 16) * (N / 64);
    int blocks = (tiles + 7) / 8;
    wmma_gemm_bf16<<<blocks, 256, 0, stream>>>(A, lda, Bt, Cin, ldcin,
                                               C, ldc, bias, M, N, K);
  };

  // ---- weight pre-pass: f32 row-major -> bf16 column-major ----
  for (int l = 0; l < 2; ++l)
    for (int d = 0; d < 2; ++d) {
      int din = (l == 0) ? HIDc : 2 * HIDc;
      convert(encWk[l][d], encWkT[l][d], din, 4 * HIDc);
      convert(encWr[l][d], encWrT[l][d], HIDc, 4 * HIDc);
    }
  for (int l = 0; l < 2; ++l) {
    int din = (l == 0) ? HIDc : H2c;
    convert(decWk[l], decWkT[l], din, 4 * H2c);
    convert(decWr[l], decWrT[l], H2c, 4 * H2c);
  }
  convert(dense_W, denseWT, H2c, VOCc);

  // ---- encoder input embedding + mask ----
  embed_mask_kernel<<<(int)(ROWS / 8), 256, 0, stream>>>(enc_tokens, emb, buf0,
                                                         maskE, (int)ROWS);
  // zero encoder initial states (fh,fc,bh,bc contiguous)
  zero_kernel<<<(4 * Bsz * HIDc) / 256, 256, 0, stream>>>(fh, 4 * Bsz * HIDc);

  // ---- encoder: 2 bidirectional layers; states carry across layers ----
  float* xin  = buf0;
  float* xout = buf1;
  int din = HIDc;
  for (int l = 0; l < 2; ++l) {
    for (int dir = 0; dir < 2; ++dir) {
      // zx = x @ Wk + b  over all timesteps  [ROWS, 1024]
      gemm(xin, din, encWkT[l][dir], nullptr, 0,
           zxBuf, 4 * HIDc, encBb[l][dir], (int)ROWS, 4 * HIDc, din);
      float* h = dir ? bh : fh;
      float* c = dir ? bc : fc;
      for (int s = 0; s < Tlen; ++s) {
        int t = dir ? (Tlen - 1 - s) : s;
        // Z = zx[:,t,:] + h @ Wr     (accumulator initialized from zx)
        gemm(h, HIDc, encWrT[l][dir],
             zxBuf + (size_t)t * 4 * HIDc, Tlen * 4 * HIDc,
             zStep, 4 * HIDc, nullptr, Bsz, 4 * HIDc, HIDc);
        lstm_pointwise_kernel<<<(Bsz * HIDc) / 256, 256, 0, stream>>>(
            zStep, h, c, maskE, t, Tlen, xout, 2 * HIDc, dir * HIDc, HIDc);
      }
    }
    float* tmp = xin; xin = xout; xout = tmp;
    din = 2 * HIDc;
  }

  // ---- decoder init: h = [fh|bh], c = [fc|bc] ----
  copy_cols_kernel<<<(Bsz * HIDc) / 256, 256, 0, stream>>>(fh, hdec, Bsz, HIDc, H2c, 0);
  copy_cols_kernel<<<(Bsz * HIDc) / 256, 256, 0, stream>>>(bh, hdec, Bsz, HIDc, H2c, HIDc);
  copy_cols_kernel<<<(Bsz * HIDc) / 256, 256, 0, stream>>>(fc, cdec, Bsz, HIDc, H2c, 0);
  copy_cols_kernel<<<(Bsz * HIDc) / 256, 256, 0, stream>>>(bc, cdec, Bsz, HIDc, H2c, HIDc);

  // ---- decoder input embedding + mask (encoder activations no longer needed) ----
  embed_mask_kernel<<<(int)(ROWS / 8), 256, 0, stream>>>(dec_tokens, emb, buf0,
                                                         maskD, (int)ROWS);

  // ---- decoder: 2 unidirectional layers, H2 = 512; h,c carry across layers ----
  float* dxin  = buf0;
  float* dxout = buf1;
  int ddin = HIDc;
  for (int l = 0; l < 2; ++l) {
    gemm(dxin, ddin, decWkT[l], nullptr, 0,
         zxBuf, 4 * H2c, decBb[l], (int)ROWS, 4 * H2c, ddin);
    for (int t = 0; t < Tlen; ++t) {
      gemm(hdec, H2c, decWrT[l],
           zxBuf + (size_t)t * 4 * H2c, Tlen * 4 * H2c,
           zStep, 4 * H2c, nullptr, Bsz, 4 * H2c, H2c);
      lstm_pointwise_kernel<<<(Bsz * H2c) / 256, 256, 0, stream>>>(
          zStep, hdec, cdec, maskD, t, Tlen, dxout, H2c, 0, H2c);
    }
    float* tmp = dxin; dxin = dxout; dxout = tmp;
    ddin = H2c;
  }

  // ---- final dense on last timestep: out[64,32000] = x[:,T-1,:] @ W + b ----
  gemm(dxin + (size_t)(Tlen - 1) * H2c, Tlen * H2c, denseWT,
       nullptr, 0, out, VOCc, dense_b, Bsz, VOCc, H2c);
}
